// RefractTracer_28690381538009
// MI455X (gfx1250) — matrix operand order repaired and checked by reading, not compiled, not run on hardware
//
#include <hip/hip_runtime.h>
#include <hip/hip_bf16.h>

// =====================================================================
// RefractTracer for MI455X (gfx1250, wave32).
//
// ~36 GFLOP of MLP work vs <1 MB of data -> compute bound. One wave per
// ray; the 64x64 hidden layer runs on v_wmma_f32_16x16x32_f16 with W2
// fragments resident in VGPRs. This revision: cap the kernel at 256
// arch VGPRs and fence the scheduler per N-tile / per M-tile so it
// stops keeping 4 accumulator octets + next-tile layer-1 live (which
// spilled into the v256+ space and caused the s_set_vgpr_msb storm).
// Latency is hidden by occupancy (thousands of independent waves), not
// intra-wave ILP.
// =====================================================================

#define N_RAYS   16384
#define HID      64
#define EPSF     1e-10f
#define IOR_AIR  1.0003f
#define IOR_OBJ  1.4723f
#define L2E_F    1.44269504088896f
#define LN2_F    0.69314718055995f

typedef __attribute__((ext_vector_type(16))) _Float16 v16h;
typedef __attribute__((ext_vector_type(8)))  _Float16 v8h;
typedef __attribute__((ext_vector_type(8)))  float    v8f;

struct f3 { float x, y, z; };

__device__ __forceinline__ float dot3(const f3 a, const f3 b) {
  return fmaf(a.x, b.x, fmaf(a.y, b.y, a.z * b.z));
}
// ---- raw-hardware transcendental helpers (single v_* trans ops) ----
__device__ __forceinline__ float fast_exp2(float x) { return __builtin_amdgcn_exp2f(x); }
__device__ __forceinline__ float fast_log2(float x) { return __builtin_amdgcn_logf(x); }
__device__ __forceinline__ float fast_rcp (float x) { return __builtin_amdgcn_rcpf(x); }
__device__ __forceinline__ float fast_rsq (float x) { return __builtin_amdgcn_rsqf(x); }

__device__ __forceinline__ float softplus_f(float x) {
  // max(x,0) + ln(1 + 2^(-|x|*log2e)) ; arg of log2 in (1,2] -> accurate.
  const float t = fast_exp2(-fabsf(x) * L2E_F);
  return fmaxf(x, 0.f) + LN2_F * fast_log2(1.f + t);
}
__device__ __forceinline__ float sigmoid_f(float x) {
  return fast_rcp(1.f + fast_exp2(-x * L2E_F));
}
__device__ __forceinline__ float tanh_f(float x) {
  // 1 - 2/(2^(2x*log2e)+1); exp2->inf gives +1, exp2->0 gives -1.
  const float e = fast_exp2(x * (2.f * L2E_F));
  return 1.f - 2.f * fast_rcp(e + 1.f);
}
// Wave-private LDS producer->consumer ordering (LDS is in-order per wave;
// this just pins the compiler's memory scheduling).
__device__ __forceinline__ void wave_sync() {
  __builtin_amdgcn_fence(__ATOMIC_ACQ_REL, "wavefront");
  __builtin_amdgcn_wave_barrier();
}
// Hard scheduling fence: keeps register pressure bounded by preventing
// the scheduler from hoisting WMMAs / pipelining across it.
__device__ __forceinline__ void sched_fence() {
  __builtin_amdgcn_sched_barrier(0);
}

// ---------------------------------------------------------------------
// Evaluate SDF at 64 samples z_j = zA + zB*j along ray (o,d).
// Writes s[0..63] into the wave-private LDS strip sS.
// W2 is consumed from resident B-fragments bfr[nt][ks] (4 N-tiles x 2 K-steps).
// ---------------------------------------------------------------------
__device__ __forceinline__ void eval_sdf64(
    const f3 o, const f3 d, const float zA, const float zB,
    const v16h bfr[4][2],
    const float* sW1, const float* sB1, const float* sB2,
    const float* sW3, const float b3v,
    float* sS, const int lane)
{
  const int mloc = lane & 15;        // this lane's M row (= its sample in tile)
  const int off  = (lane >> 4) << 3; // A-fragment K sub-offset: 0 or 8

  for (int t = 0; t < 4; ++t) {      // 4 M-tiles of 16 samples
    const int   j  = t * 16 + mloc;
    const float zj = fmaf(zB, (float)j, zA);
    const f3 p = { fmaf(zj, d.x, o.x), fmaf(zj, d.y, o.y), fmaf(zj, d.z, o.z) };

    // -------- layer 1 (K=3) straight into WMMA A-fragment layout -----
    // 16-bit A 16x32: lane<16 holds K in {off..off+7, off+16..off+23}.
    v16h a0, a1;
#pragma unroll
    for (int rg = 0; rg < 4; ++rg) {
      const int kb = off + (rg & 1) * 16 + (rg >> 1) * 32;
      const float4 wx0 = *(const float4*)(sW1 + kb);
      const float4 wx1 = *(const float4*)(sW1 + kb + 4);
      const float4 wy0 = *(const float4*)(sW1 + 64 + kb);
      const float4 wy1 = *(const float4*)(sW1 + 64 + kb + 4);
      const float4 wz0 = *(const float4*)(sW1 + 128 + kb);
      const float4 wz1 = *(const float4*)(sW1 + 128 + kb + 4);
      const float4 bb0 = *(const float4*)(sB1 + kb);
      const float4 bb1 = *(const float4*)(sB1 + kb + 4);
      const float wx[8] = {wx0.x, wx0.y, wx0.z, wx0.w, wx1.x, wx1.y, wx1.z, wx1.w};
      const float wy[8] = {wy0.x, wy0.y, wy0.z, wy0.w, wy1.x, wy1.y, wy1.z, wy1.w};
      const float wz[8] = {wz0.x, wz0.y, wz0.z, wz0.w, wz1.x, wz1.y, wz1.z, wz1.w};
      const float bv[8] = {bb0.x, bb0.y, bb0.z, bb0.w, bb1.x, bb1.y, bb1.z, bb1.w};
#pragma unroll
      for (int e = 0; e < 8; ++e) {
        const float zv = fmaf(p.x, wx[e], fmaf(p.y, wy[e], fmaf(p.z, wz[e], bv[e])));
        const _Float16 h = (_Float16)softplus_f(zv);
        const int el = (rg & 1) * 8 + e;
        if ((rg >> 1) == 0) a0[el] = h; else a1[el] = h;
      }
    }
    sched_fence();   // layer-1 complete before the WMMA cluster

    // -------- layer 2 + 3 fused per N-tile: ONE v8f accumulator ------
    // C/D layout: lane holds column n = nt*16 + (lane&15),
    // rows m = (lane>=16 ? 8 : 0) + r for r in 0..7.
    float pm[8];
#pragma unroll
    for (int r = 0; r < 8; ++r) pm[r] = 0.f;
#pragma unroll
    for (int nt = 0; nt < 4; ++nt) {
      v8f c = {};
      c = __builtin_amdgcn_wmma_f32_16x16x32_f16(false, a0, false, bfr[nt][0],
                                                 (short)0, c, false, false);
      c = __builtin_amdgcn_wmma_f32_16x16x32_f16(false, a1, false, bfr[nt][1],
                                                 (short)0, c, false, false);
      const int   n   = nt * 16 + mloc;
      const float b2n = sB2[n];
      const float w3n = sW3[n];
#pragma unroll
      for (int r = 0; r < 8; ++r)
        pm[r] = fmaf(softplus_f(c[r] + b2n), w3n, pm[r]);
      sched_fence(); // retire this N-tile's accumulator before the next
    }
#pragma unroll
    for (int m = 1; m < 16; m <<= 1) {
#pragma unroll
      for (int r = 0; r < 8; ++r) pm[r] += __shfl_xor(pm[r], m, 32);
    }

    // Writer lanes chosen so the lane's own point index equals the D-row:
    //  m in 0..7  -> lane m       (r = m)
    //  m in 8..15 -> lane 16 + m  (r = m - 8), i.e. lanes 24..31
    int wr = -1;
    if (lane < 8)        wr = lane;
    else if (lane >= 24) wr = lane - 24;
    float mv = 0.f;
#pragma unroll
    for (int r = 0; r < 8; ++r) mv = (r == wr) ? pm[r] : mv;
    if (wr >= 0) {
      const float pp = fmaxf(dot3(p, p), EPSF);
      const float pn = pp * fast_rsq(pp);       // sqrt via rsq*x
      sS[j] = pn - 0.5f + 0.02f * tanh_f(mv + b3v);
    }
    sched_fence();   // no software pipelining across M-tiles
  }
}

// ---------------------------------------------------------------------
// Find first sign crossing among 64 samples via wave ballots.
// ---------------------------------------------------------------------
__device__ __forceinline__ void bracket64(
    const float* sS, const int lane, const bool enter,
    const float zA, const float zB,
    bool& valid, float& z0, float& z1, float& s0, float& s1)
{
  const float sj  = sS[lane];
  const float sj1 = sS[lane + 1];
  const float sk  = sS[lane + 31];
  const float sk1 = sS[lane + 32];
  const bool c0 = (lane <= 30) &&
      (enter ? (sj > 0.f && sj1 <= 0.f) : (sj < 0.f && sj1 >= 0.f));
  const bool c1 =
      (enter ? (sk > 0.f && sk1 <= 0.f) : (sk < 0.f && sk1 >= 0.f));
  const unsigned m0 = (unsigned)__ballot(c0);
  const unsigned m1 = (unsigned)__ballot(c1);
  valid = (m0 | m1) != 0u;
  int idx = 0;
  if (m0)      idx = __builtin_ctz(m0);
  else if (m1) idx = 31 + __builtin_ctz(m1);
  z0 = fmaf(zB, (float)idx, zA);
  z1 = fmaf(zB, (float)(idx + 1), zA);
  s0 = sS[idx];
  s1 = sS[idx + 1];
}

// ---------------------------------------------------------------------
// Unit normal of SDF at p: analytic backprop through the tiny MLP,
// data-parallel over hidden units (2 per lane), LDS broadcasts.
// ---------------------------------------------------------------------
__device__ __forceinline__ f3 sdf_grad(
    const f3 p,
    const float* sW1, const float* sB1, const _Float16* sW2t,
    const float* sB2, const float* sW3, const float b3v,
    float* hbuf, float* ubuf, const int lane)
{
  float sg1[2];
#pragma unroll
  for (int u = 0; u < 2; ++u) {
    const int k = lane * 2 + u;
    const float z = fmaf(p.x, sW1[k],
                    fmaf(p.y, sW1[64 + k],
                    fmaf(p.z, sW1[128 + k], sB1[k])));
    hbuf[k] = softplus_f(z);
    sg1[u]  = sigmoid_f(z);
  }
  wave_sync();

  float mlp = 0.f;
#pragma unroll
  for (int u = 0; u < 2; ++u) {
    const int n = lane * 2 + u;
    float z = sB2[n];
#pragma unroll 8
    for (int k = 0; k < HID; ++k)
      z = fmaf(hbuf[k], (float)sW2t[n * HID + k], z);
    mlp = fmaf(softplus_f(z), sW3[n], mlp);
    ubuf[n] = sigmoid_f(z) * sW3[n];
  }
  wave_sync();
#pragma unroll
  for (int m = 1; m < 32; m <<= 1) mlp += __shfl_xor(mlp, m, 32);
  mlp += b3v;

  float vv[2];
#pragma unroll
  for (int u = 0; u < 2; ++u) {
    const int k = lane * 2 + u;
    float acc = 0.f;
#pragma unroll 8
    for (int n = 0; n < HID; ++n)
      acc = fmaf((float)sW2t[n * HID + k], ubuf[n], acc);
    vv[u] = sg1[u] * acc;
  }
  wave_sync();
  hbuf[lane * 2 + 0] = vv[0];
  hbuf[lane * 2 + 1] = vv[1];
  wave_sync();

  const float th = tanh_f(mlp);
  const float gm = 0.02f * (1.f - th * th);
  float gx = 0.f, gy = 0.f, gz = 0.f;
#pragma unroll 8
  for (int k = 0; k < HID; ++k) {
    const float vk = hbuf[k];
    gx = fmaf(sW1[k], vk, gx);
    gy = fmaf(sW1[64 + k], vk, gy);
    gz = fmaf(sW1[128 + k], vk, gz);
  }
  const float ipn = fast_rsq(fmaxf(dot3(p, p), EPSF));
  f3 g = { fmaf(gm, gx, p.x * ipn),
           fmaf(gm, gy, p.y * ipn),
           fmaf(gm, gz, p.z * ipn) };
  const float ign = fast_rsq(fmaxf(dot3(g, g), EPSF));
  return { g.x * ign, g.y * ign, g.z * ign };
}

// ---------------------------------------------------------------------
// Full coarse+fine intersection: depth + unit normal + validity.
// ---------------------------------------------------------------------
__device__ __forceinline__ void intersect(
    const f3 o, const f3 d, const bool enter,
    const v16h bfr[4][2],
    const float* sW1, const float* sB1, const _Float16* sW2t,
    const float* sB2, const float* sW3, const float b3v,
    float* sS, float* hbuf, float* ubuf, const int lane,
    bool& valid, float& depth, f3& nrm)
{
  const float aa  = dot3(d, d);
  const float bb  = 2.f * dot3(o, d);
  const float mid = -0.5f * bb * fast_rcp(aa);
  const float zn  = fmaxf(mid - 1.f, 0.f);
  const float zf  = mid + 1.f;

  const float zA = zn, zB = (zf - zn) * (1.f / 63.f);
  eval_sdf64(o, d, zA, zB, bfr, sW1, sB1, sB2, sW3, b3v, sS, lane);
  wave_sync();
  bool v1; float z0, z1, s0, s1;
  bracket64(sS, lane, enter, zA, zB, v1, z0, z1, s0, s1);
  wave_sync();

  const float zA2 = z0, zB2 = (z1 - z0) * (1.f / 63.f);
  eval_sdf64(o, d, zA2, zB2, bfr, sW1, sB1, sB2, sW3, b3v, sS, lane);
  wave_sync();
  bool v2; float z0f, z1f, s0f, s1f;
  bracket64(sS, lane, enter, zA2, zB2, v2, z0f, z1f, s0f, s1f);

  valid = v1 && v2;
  float den = s0f - s1f;
  den = (fabsf(den) < EPSF) ? EPSF : den;
  depth = fmaf(z1f - z0f, s0f * fast_rcp(den), z0f);
  depth = valid ? depth : 1.0f;

  const f3 p = { fmaf(depth, d.x, o.x),
                 fmaf(depth, d.y, o.y),
                 fmaf(depth, d.z, o.z) };
  nrm = sdf_grad(p, sW1, sB1, sW2t, sB2, sW3, b3v, hbuf, ubuf, lane);
}

// ---------------------------------------------------------------------
// Snell refraction + Fresnel attenuation (all lanes redundant scalar).
// ---------------------------------------------------------------------
__device__ __forceinline__ void refract_dir(
    const f3 l, const f3 n, const float e1, const float e2,
    f3& t, float& att)
{
  const float ct = -dot3(l, n);
  const f3 ip = { fmaf(ct, n.x, l.x), fmaf(ct, n.y, l.y), fmaf(ct, n.z, l.z) };
  const float sc = e1 * fast_rcp(e2);
  const f3 tp = { sc * ip.x, sc * ip.y, sc * ip.z };
  const float tpn = fminf(fmaxf(dot3(tp, tp), 0.f), 0.999999f);
  const float om  = fmaxf(1.f - tpn, EPSF);
  const float ti  = om * fast_rsq(om);          // sqrt(1 - tpn)
  f3 tt = { fmaf(-ti, n.x, tp.x), fmaf(-ti, n.y, tp.y), fmaf(-ti, n.z, tp.z) };
  const float itn = fast_rsq(fmaxf(dot3(tt, tt), EPSF));
  t = { tt.x * itn, tt.y * itn, tt.z * itn };
  const float ctt = -dot3(t, n);
  const float ei = (ctt * e1 - ct * e2) * fast_rcp(fmaxf(ctt * e1 + ct * e2, EPSF));
  const float ep = (ct * e1 - ctt * e2) * fast_rcp(fmaxf(ct * e1 + ctt * e2, EPSF));
  att = fminf(fmaxf(0.5f * (ei * ei + ep * ep), 0.f), 1.f);
}

__device__ __forceinline__ f3 reflect_dir(const f3 l, const f3 n)
{
  const float ct = -dot3(l, n);
  f3 r = { fmaf(2.f * ct, n.x, l.x),
           fmaf(2.f * ct, n.y, l.y),
           fmaf(2.f * ct, n.z, l.z) };
  const float irn = fast_rsq(fmaxf(dot3(r, r), EPSF));
  return { r.x * irn, r.y * irn, r.z * irn };
}

// =====================================================================
// Kernel: 128 threads = 4 waves = 4 rays per block.
// Cap at 256 arch VGPRs: avoids the v256+ addressing mode entirely
// (no s_set_vgpr_msb) and allows >=4 waves/SIMD of occupancy.
// =====================================================================
__global__ __launch_bounds__(128)
__attribute__((amdgpu_num_vgpr(256)))
void refract_tracer_kernel(
    const float* __restrict__ rays_o, const float* __restrict__ rays_d,
    const float* __restrict__ W1, const float* __restrict__ b1,
    const float* __restrict__ W2, const float* __restrict__ b2,
    const float* __restrict__ W3, const float* __restrict__ b3,
    float* __restrict__ out)
{
  __shared__ __attribute__((aligned(32))) _Float16 sW2t[HID * HID]; // [n][k], f16
  __shared__ __attribute__((aligned(16))) float sW1[3 * HID];
  __shared__ __attribute__((aligned(16))) float sB1[HID];
  __shared__ __attribute__((aligned(16))) float sB2[HID];
  __shared__ __attribute__((aligned(16))) float sW3[HID];
  __shared__ float sB3;
  __shared__ __attribute__((aligned(16))) float sS[4][64];   // per-wave SDF strip
  __shared__ __attribute__((aligned(16))) float sHB[4][64];  // per-wave grad scratch
  __shared__ __attribute__((aligned(16))) float sUB[4][64];  // per-wave grad scratch

  const int tid = threadIdx.x;

  // Stage weights: W2 (f32, row-major [k][n]) -> transposed f16 [n][k] in LDS.
  __builtin_prefetch(W2, 0, 3);
  for (int i = tid; i < HID * HID; i += 128) {
    const int k = i >> 6, n = i & 63;
    sW2t[n * HID + k] = (_Float16)W2[i];
  }
  for (int i = tid; i < 3 * HID; i += 128) sW1[i] = W1[i];
  if (tid < HID) { sB1[tid] = b1[tid]; sB2[tid] = b2[tid]; sW3[tid] = W3[tid]; }
  if (tid == 0) sB3 = b3[0];
  __syncthreads();

  const int wv   = tid >> 5;
  const int lane = tid & 31;
  const int ray  = blockIdx.x * 4 + wv;
  const float b3v = sB3;

  // Ray data (broadcast loads).
  const f3 o = { rays_o[ray * 3 + 0], rays_o[ray * 3 + 1], rays_o[ray * 3 + 2] };
  const f3 d = { rays_d[ray * 3 + 0], rays_d[ray * 3 + 1], rays_d[ray * 3 + 2] };

  // Load the 8 resident W2 B-fragments (table K order, matching A).
  // B 32x16 f16 layout: lane n = lane&15; lane half h holds K {8h..8h+7, 16+8h..16+8h+7}.
  v16h bfr[4][2];
#pragma unroll
  for (int nt = 0; nt < 4; ++nt) {
#pragma unroll
    for (int ks = 0; ks < 2; ++ks) {
      const int n  = nt * 16 + (lane & 15);
      const int kb = ks * 32 + ((lane >> 4) << 3);
      const v8h* pv = (const v8h*)&sW2t[n * HID + kb];
      const v8h lo = pv[0];
      const v8h hi = pv[2];
      v16h bb;
#pragma unroll
      for (int e = 0; e < 8; ++e) { bb[e] = lo[e]; bb[e + 8] = hi[e]; }
      bfr[nt][ks] = bb;
    }
  }

  float* wsS = sS[wv];
  float* wHB = sHB[wv];
  float* wUB = sUB[wv];

  // ---- first intersection (air -> object) ----
  bool v1; float depth1; f3 n1;
  intersect(o, d, true, bfr, sW1, sB1, sW2t, sB2, sW3, b3v,
            wsS, wHB, wUB, lane, v1, depth1, n1);
  const f3 fsp = { fmaf(depth1, d.x, o.x),
                   fmaf(depth1, d.y, o.y),
                   fmaf(depth1, d.z, o.z) };
  f3 refr1; float att1;
  refract_dir(d, n1, IOR_AIR, IOR_OBJ, refr1, att1);
  const f3 refl1 = reflect_dir(d, n1);

  // ---- second intersection (object -> air) ----
  bool v2; float depth2; f3 n2;
  intersect(fsp, refr1, false, bfr, sW1, sB1, sW2t, sB2, sW3, b3v,
            wsS, wHB, wUB, lane, v2, depth2, n2);
  const f3 ssp = { fmaf(depth2, refr1.x, fsp.x),
                   fmaf(depth2, refr1.y, fsp.y),
                   fmaf(depth2, refr1.z, fsp.z) };
  const f3 n2neg = { -n2.x, -n2.y, -n2.z };
  f3 refr2; float att2;
  refract_dir(refr1, n2neg, IOR_OBJ, IOR_AIR, refr2, att2);

  const bool fin = v1 && v2;

  const f3 roo = fin ? ssp   : o;
  const f3 rdo = fin ? refr2 : d;
  const f3 ror = fin ? fsp   : o;
  const f3 rdr = fin ? refl1 : d;
  const float a1o = fin ? att1 : 0.f;
  const float a2o = fin ? att2 : 0.f;

  // Outputs, concatenated flat in reference return order.
  if (lane < 3) {
    const float c0 = (lane == 0) ? roo.x : (lane == 1) ? roo.y : roo.z;
    const float c1 = (lane == 0) ? rdo.x : (lane == 1) ? rdo.y : rdo.z;
    const float c2 = (lane == 0) ? ror.x : (lane == 1) ? ror.y : ror.z;
    const float c3 = (lane == 0) ? rdr.x : (lane == 1) ? rdr.y : rdr.z;
    out[0 * 3 * N_RAYS + ray * 3 + lane] = c0;
    out[1 * 3 * N_RAYS + ray * 3 + lane] = c1;
    out[2 * 3 * N_RAYS + ray * 3 + lane] = c2;
    out[3 * 3 * N_RAYS + ray * 3 + lane] = c3;
  }
  if (lane == 0) {
    out[12 * N_RAYS + ray] = a1o;
    out[13 * N_RAYS + ray] = a2o;
    out[14 * N_RAYS + ray] = fin ? 1.f : 0.f;
  }
}

extern "C" void kernel_launch(void* const* d_in, const int* in_sizes, int n_in,
                              void* d_out, int out_size, void* d_ws, size_t ws_size,
                              hipStream_t stream) {
  (void)in_sizes; (void)n_in; (void)out_size; (void)d_ws; (void)ws_size;
  const float* rays_o = (const float*)d_in[0];
  const float* rays_d = (const float*)d_in[1];
  const float* W1     = (const float*)d_in[2];
  const float* b1     = (const float*)d_in[3];
  const float* W2     = (const float*)d_in[4];
  const float* b2     = (const float*)d_in[5];
  const float* W3     = (const float*)d_in[6];
  const float* b3     = (const float*)d_in[7];
  float* out = (float*)d_out;

  refract_tracer_kernel<<<N_RAYS / 4, 128, 0, stream>>>(
      rays_o, rays_d, W1, b1, W2, b2, W3, b3, out);
}
